// AlignConv_40553081208979
// MI455X (gfx1250) — compile-verified, matrix operand order checked by reading, not run on hardware
//
#include <hip/hip_runtime.h>
#include <hip/hip_bf16.h>

typedef __attribute__((ext_vector_type(16))) _Float16 v16h;
typedef __attribute__((ext_vector_type(8)))  _Float16 v8h;
typedef __attribute__((ext_vector_type(8)))  float    v8f;

#define B_     8
#define CIN    256
#define COUT   256
#define H_     64
#define W_     64
#define K2     9
#define NSTEPS 72   // K dim: 9 taps * 8 chunks of 32 channels
#define CHUNK  24   // K-steps staged in LDS per pipeline stage
#define NTILE  32   // spatial positions (N) per block = 2 WMMA N-fragments

// ---------------------------------------------------------------------------
// Prolog 1: x [B,Cin,H,W] f32  ->  xT [B,H,W,Cin] f16   (channel-contiguous)
// ---------------------------------------------------------------------------
__global__ __launch_bounds__(256)
void k_xpose(const float* __restrict__ x, _Float16* __restrict__ xT) {
    size_t i = (size_t)blockIdx.x * 256 + threadIdx.x;  // ((b*H+h)*W+w)*CIN + c
    int c = (int)(i & 255);
    size_t p = i >> 8;
    int w = (int)(p & 63); p >>= 6;
    int h = (int)(p & 63);
    int b = (int)(p >> 6);
    xT[i] = (_Float16)x[(((size_t)b * CIN + c) * H_ + h) * W_ + w];
}

// ---------------------------------------------------------------------------
// Prolog 2: weight [Cout,Cin,3,3] f32 -> WMMA A-fragment order f16
//   wSwz[mt][s][lane][e],  mt=0..15 (16 Cout each), s=0..71 K-steps
//   16-bit A 16x32 layout: m = lane&15,
//   K-in-step d = (e<8 ? e : e+8) + (lane<16 ? 0 : 8)
// ---------------------------------------------------------------------------
__global__ __launch_bounds__(256)
void k_wswz(const float* __restrict__ wgt, _Float16* __restrict__ wSwz) {
    int j = blockIdx.x * 256 + threadIdx.x;   // 16*72*32*16 elements
    int e = j & 15;
    int t = j >> 4;
    int lane = t & 31; t >>= 5;
    int s  = t % NSTEPS;
    int mt = t / NSTEPS;
    int m  = mt * 16 + (lane & 15);
    int d  = (e < 8 ? e : e + 8) + (lane < 16 ? 0 : 8);
    int Kg = s * 32 + d;          // global K = k*256 + c
    int k  = Kg >> 8;
    int c  = Kg & 255;
    wSwz[j] = (_Float16)wgt[((size_t)m * CIN + c) * K2 + k];
}

// ---------------------------------------------------------------------------
// Main fused kernel. Block: 256 threads (8 waves), 32 pixels x 256 Cout.
// Each wave: 2 M-tiles x 2 N-tiles -> 4 x v_wmma_f32_16x16x32_f16 per K-step,
// reusing each A fragment for both N-tiles (halves weight traffic).
// ---------------------------------------------------------------------------
__global__ __launch_bounds__(256)
void k_alignconv(const float* __restrict__ anchors,
                 const _Float16* __restrict__ xT,
                 const _Float16* __restrict__ wSwz,
                 float* __restrict__ out) {
    __shared__ _Float16 sB[CHUNK * 2 * 32 * 16];  // 49.2 KB: [step][nt][lane][16]
    __shared__ float sWy[NTILE * K2], sWx[NTILE * K2];
    __shared__ int   sY0[NTILE * K2], sX0[NTILE * K2];

    const int tid = threadIdx.x;
    const int w0  = blockIdx.x * NTILE;
    const int h   = blockIdx.y;
    const int b   = blockIdx.z;

    // ---- Stage 1: per (pixel n, tap k) absolute sample position ----------
    for (int pi = tid; pi < NTILE * K2; pi += 256) {
        int n = pi / K2, k = pi % K2;
        const float* a = anchors + ((size_t)b * (H_ * W_) + (size_t)h * W_ + (w0 + n)) * 5;
        const float invS = 1.0f / 8.0f;
        float xc = a[0] * invS, yc = a[1] * invS;
        float dw = a[2] * (invS / 3.0f), dh = a[3] * (invS / 3.0f);
        float ang = a[4];
        float sn = __sinf(ang), cs = __cosf(ang);
        float dx = (float)(k % 3 - 1), dy = (float)(k / 3 - 1);
        float xk = dw * dx, yk = dh * dy;
        float px = cs * xk - sn * yk + xc;   // conv-grid terms cancel:
        float py = sn * xk + cs * yk + yc;   // absolute sampling position
        float fy = floorf(py), fx = floorf(px);
        sY0[pi] = (int)fy;  sX0[pi] = (int)fx;
        sWy[pi] = py - fy;  sWx[pi] = px - fx;
    }

    v8f acc00 = {}, acc01 = {}, acc10 = {}, acc11 = {};
    const int wv = tid >> 5, lane = tid & 31;
    const int mt0 = 2 * wv, mt1 = 2 * wv + 1;

    for (int s0 = 0; s0 < NSTEPS; s0 += CHUNK) {
        __syncthreads();   // protect sB from previous round's readers

        // ---- Phase A: bilinear-sample B-panel into LDS (fragment layout,
        //      packed f16 math -> v_pk_fma_f16) ----------------------------
        #pragma unroll
        for (int i = 0; i < (CHUNK * 2 * 32) / 256; ++i) {
            int j   = tid + 256 * i;       // lane-fragment index in panel
            int L   = j & 31;              // fragment lane slot
            int t   = j >> 5;
            int nt  = t & 1;               // which N fragment (16 pixels)
            int ssl = t >> 1;              // local K-step
            int s   = s0 + ssl;
            int k   = s >> 3;              // tap
            int cc  = (s & 7) * 32;        // channel chunk base
            int n   = nt * 16 + (L & 15);  // pixel within block tile
            int cb  = cc + ((L < 16) ? 0 : 8);   // first run of 8 channels
            int pi  = n * K2 + k;
            int y0 = sY0[pi], x0 = sX0[pi];
            float wy1 = sWy[pi], wx1 = sWx[pi];
            float wq[4] = { (1.f - wy1) * (1.f - wx1), (1.f - wy1) * wx1,
                            wy1 * (1.f - wx1),         wy1 * wx1 };
            v8h acc0h = {};   // channels cb..cb+7      -> frag elems 0..7
            v8h acc1h = {};   // channels cb+16..cb+23  -> frag elems 8..15
            #pragma unroll
            for (int cr = 0; cr < 4; ++cr) {
                int y = y0 + (cr >> 1);
                int x = x0 + (cr & 1);
                if (y >= 0 && y < H_ && x >= 0 && x < W_) {
                    const _Float16* p =
                        xT + ((((size_t)b * H_ + y) * W_ + x) * CIN + cb);
                    v8h r0 = *(const v8h*)p;
                    v8h r1 = *(const v8h*)(p + 16);
                    _Float16 gh = (_Float16)wq[cr];
                    v8h gv;
                    #pragma unroll
                    for (int q = 0; q < 8; ++q) gv[q] = gh;
                    acc0h += gv * r0;      // v_pk_fma_f16
                    acc1h += gv * r1;
                }
            }
            _Float16* dst = &sB[((ssl * 2 + nt) * 32 + L) * 16];
            *(v8h*)dst       = acc0h;
            *(v8h*)(dst + 8) = acc1h;
        }
        __syncthreads();

        // ---- Phase B: WMMA over staged panel (uniform flow, EXEC all 1s) --
        #pragma unroll 2
        for (int ss = 0; ss < CHUNK; ++ss) {
            int s = s0 + ss;
            v16h bf0 = *(const v16h*)&sB[((ss * 2 + 0) * 32 + lane) * 16];
            v16h bf1 = *(const v16h*)&sB[((ss * 2 + 1) * 32 + lane) * 16];
            const _Float16* pa0 =
                wSwz + ((((size_t)mt0 * NSTEPS) + s) * 32 + lane) * 16;
            const _Float16* pa1 =
                wSwz + ((((size_t)mt1 * NSTEPS) + s) * 32 + lane) * 16;
            v16h a0 = *(const v16h*)pa0;
            v16h a1 = *(const v16h*)pa1;
            __builtin_prefetch((const void*)(pa0 + 4 * 512), 0, 1);
            __builtin_prefetch((const void*)(pa1 + 4 * 512), 0, 1);
            acc00 = __builtin_amdgcn_wmma_f32_16x16x32_f16(
                        false, a0, false, bf0, (short)0, acc00, false, false);
            acc01 = __builtin_amdgcn_wmma_f32_16x16x32_f16(
                        false, a0, false, bf1, (short)0, acc01, false, false);
            acc10 = __builtin_amdgcn_wmma_f32_16x16x32_f16(
                        false, a1, false, bf0, (short)0, acc10, false, false);
            acc11 = __builtin_amdgcn_wmma_f32_16x16x32_f16(
                        false, a1, false, bf1, (short)0, acc11, false, false);
        }
    }

    // ---- Epilogue: ReLU + store. D layout: lane holds col n = lane&15,
    //      VGPR r holds row m = r + 8*(lane/16). Lanes contiguous in w. -----
    int nlo = lane & 15;
    int mo  = (lane >> 4) << 3;
    #pragma unroll
    for (int r = 0; r < 8; ++r) {
        int o0 = mt0 * 16 + r + mo;
        int o1 = mt1 * 16 + r + mo;
        float v00 = acc00[r]; v00 = v00 > 0.f ? v00 : 0.f;
        float v01 = acc01[r]; v01 = v01 > 0.f ? v01 : 0.f;
        float v10 = acc10[r]; v10 = v10 > 0.f ? v10 : 0.f;
        float v11 = acc11[r]; v11 = v11 > 0.f ? v11 : 0.f;
        size_t base0 = (((size_t)b * COUT + o0) * H_ + h) * W_ + w0;
        size_t base1 = (((size_t)b * COUT + o1) * H_ + h) * W_ + w0;
        out[base0 + nlo]      = v00;
        out[base0 + 16 + nlo] = v01;
        out[base1 + nlo]      = v10;
        out[base1 + 16 + nlo] = v11;
    }
}

// ---------------------------------------------------------------------------
extern "C" void kernel_launch(void* const* d_in, const int* in_sizes, int n_in,
                              void* d_out, int out_size, void* d_ws, size_t ws_size,
                              hipStream_t stream) {
    const float* x       = (const float*)d_in[0];
    const float* anchors = (const float*)d_in[1];
    const float* wgt     = (const float*)d_in[2];
    // d_in[3..5] = feat_h, feat_w, stride (compile-time constants 64,64,8)

    _Float16* xT   = (_Float16*)d_ws;                                   // 16.78 MB
    _Float16* wSwz = (_Float16*)((char*)d_ws +
                       (size_t)B_ * H_ * W_ * CIN * sizeof(_Float16)); // +1.18 MB

    float* out = (float*)d_out;

    k_xpose<<<(B_ * H_ * W_ * CIN) / 256, 256, 0, stream>>>(x, xT);
    k_wswz<<<(16 * NSTEPS * 32 * 16) / 256, 256, 0, stream>>>(wgt, wSwz);

    dim3 grid(W_ / NTILE, H_, B_);
    k_alignconv<<<grid, 256, 0, stream>>>(anchors, xT, wSwz, out);
}
